// Sequence_52123723104803
// MI455X (gfx1250) — matrix operand order repaired and checked by reading, not compiled, not run on hardware
//
#include <hip/hip_runtime.h>
#include <hip/hip_bf16.h>

// ---------------------------------------------------------------------------
// 2-layer LSTM, batch-parallel persistent-state design for MI455X (gfx1250).
//  - 32 blocks x 512 threads (16 waves, wave32). Block b owns batch rows
//    [16b, 16b+16) and iterates all T timesteps (recurrence is independent
//    across rows -> zero grid-level synchronization).
//  - Wave w owns hidden units [16w,16w+16): computes i/f/g/o 16x16 tiles for
//    those units via v_wmma_f32_16x16x32_bf16, keeps c-state in VGPRs across
//    all timesteps, does the LSTM nonlinearity fully in registers using the
//    native gfx1250 v_tanh_f32 transcendental (branchless, co-executes).
//  - A = [x | h] lives in LDS as bf16 (16x512 per layer, 16KB each).
//  - Weights packed once into bf16 WMMA B-fragment layout -> one contiguous
//    32B load per lane per fragment, streamed from L2 (2MB total, L2-resident).
// ---------------------------------------------------------------------------

typedef __attribute__((ext_vector_type(16))) __bf16 v16bf;
typedef __attribute__((ext_vector_type(8)))  float  v8f;

#define HID     256
#define SEG     32
#define HIST    8
#define BATCH   512
#define ROWS    16            // batch rows per block
#define NBLK    (BATCH/ROWS)  // 32 blocks
#define THREADS 512           // 16 waves
#define KDIM    512           // K of gate GEMM: [x|h] or [h1|h2]
#define KT      16            // K tiles of 32
#define PW_ELEMS (64*16*32*16)  // per-layer packed weight elems (524288)

union FragU { uint4 q[2]; v16bf v; };

// --- fast activations: everything routed through native v_tanh_f32 ---------
#if __has_builtin(__builtin_amdgcn_tanhf)
__device__ __forceinline__ float fast_tanh(float x) {
  return __builtin_amdgcn_tanhf(x);
}
__device__ __forceinline__ float sigf(float x) {
  return __builtin_amdgcn_tanhf(x * 0.5f) * 0.5f + 0.5f;
}
#else
__device__ __forceinline__ float fast_tanh(float x) {
  // branchless: tanh(x) = sign(x) * (1 - e) / (1 + e), e = exp(-2|x|)
  float e = __expf(-2.0f * __builtin_fabsf(x));
  float r = (1.0f - e) * __builtin_amdgcn_rcpf(1.0f + e);
  return __builtin_copysignf(r, x);
}
__device__ __forceinline__ float sigf(float x) {
  return __builtin_amdgcn_rcpf(1.0f + __expf(-x));
}
#endif

// --- Prep: pack [w_ih | w_hh] (fp32, row-major 1024x256 each) into bf16
// WMMA B-fragments: frag(ntile,ktile)[lane][e], where
//   n = ntile*16 + (lane&15),  k = ktile*32 + (lane&16 ? 16 : 0) + e
// Also precompute bias sums b_ih + b_hh.
__global__ void pack_kernel(const float* __restrict__ w_ih1, const float* __restrict__ w_hh1,
                            const float* __restrict__ b_ih1, const float* __restrict__ b_hh1,
                            const float* __restrict__ w_ih2, const float* __restrict__ w_hh2,
                            const float* __restrict__ b_ih2, const float* __restrict__ b_hh2,
                            __bf16* __restrict__ pW1, __bf16* __restrict__ pW2,
                            float* __restrict__ bias1, float* __restrict__ bias2)
{
  unsigned gid   = blockIdx.x * 256u + threadIdx.x;   // [0, 2^20)
  unsigned layer = gid >> 19;
  unsigned idx   = gid & 0x7FFFFu;                    // [0, 2^19) per layer
  unsigned e     = idx & 15u;
  unsigned lane  = (idx >> 4) & 31u;
  unsigned kt    = (idx >> 9) & 15u;
  unsigned nt    = idx >> 13;                         // 0..63

  int n = (int)(nt * 16u + (lane & 15u));
  int k = (int)(kt * 32u + ((lane & 16u) ? 16u : 0u) + e);
  const float* wi = layer ? w_ih2 : w_ih1;
  const float* wh = layer ? w_hh2 : w_hh1;
  float v = (k < HID) ? wi[n * HID + k] : wh[n * HID + (k - HID)];
  (layer ? pW2 : pW1)[idx] = (__bf16)v;

  if (gid < 2048u) {
    int l2 = (int)(gid >> 10), nn = (int)(gid & 1023u);
    if (l2) bias2[nn] = b_ih2[nn] + b_hh2[nn];
    else    bias1[nn] = b_ih1[nn] + b_hh1[nn];
  }
}

// One gate GEMM: acc[g] (g = i,f,g,o tile for this wave's 16 units)
//   = bias + A(16x512,bf16 in LDS) x W^T slice via 16x16x32 bf16 WMMA.
__device__ __forceinline__ void gemm_gates(const __bf16 (*A)[KDIM],
                                           const __bf16* __restrict__ pW,
                                           const float bv[4],
                                           int wave, int lane, v8f acc[4])
{
  const int lane15 = lane & 15;
  const int aoff   = (lane & 16) ? 16 : 0;  // A-frag: lane<16 -> K{0-15}, else K{16-31}
  #pragma unroll
  for (int g = 0; g < 4; ++g) {
    #pragma unroll
    for (int e = 0; e < 8; ++e) acc[g][e] = bv[g];
  }
  for (int kt = 0; kt < KT; ++kt) {
    FragU af;
    af.q[0] = *(const uint4*)&A[lane15][kt * 32 + aoff];
    af.q[1] = *(const uint4*)&A[lane15][kt * 32 + 8 + aoff];
    #pragma unroll
    for (int g = 0; g < 4; ++g) {
      FragU bfr;
      const __bf16* bp = pW + ((((g * 16 + wave) * 16 + kt) * 32 + lane) << 4);
      bfr.q[0] = *(const uint4*)(bp);
      bfr.q[1] = *(const uint4*)(bp + 8);
      acc[g] = __builtin_amdgcn_wmma_f32_16x16x32_bf16(
          false, af.v, false, bfr.v, (short)0, acc[g], false, false);
    }
  }
}

__global__ __launch_bounds__(THREADS, 1)
void lstm_main(const float* __restrict__ input,
               const __bf16* __restrict__ pW1, const __bf16* __restrict__ pW2,
               const float* __restrict__ bias1, const float* __restrict__ bias2,
               const float* __restrict__ w_lin, const float* __restrict__ b_lin,
               float* __restrict__ out, int T, int NP)
{
  __shared__ __bf16 A1[ROWS][KDIM];   // [x (0..255) | h1 (256..511)]
  __shared__ __bf16 A2[ROWS][KDIM];   // [h1 (0..255) | h2 (256..511)]
  __shared__ float  wl[HID];
  __shared__ float  red[16][16];

  const int tid    = (int)threadIdx.x;
  const int wave   = tid >> 5;
  const int lane   = tid & 31;
  const int lane15 = lane & 15;
  const int hiHalf = (lane >> 4) & 1;   // C/D layout: lanes 16-31 hold rows M=8..15
  const int r0     = (int)blockIdx.x * ROWS;

  for (int i = tid; i < HID; i += THREADS) wl[i] = w_lin[i];
  for (int i = tid; i < ROWS * HID; i += THREADS) {
    int r = i >> 8, c = i & 255;
    A1[r][HID + c] = (__bf16)0.0f;    // h1 = 0
    A2[r][HID + c] = (__bf16)0.0f;    // h2 = 0
  }
  const float blin = b_lin[0];

  float bv1[4], bv2[4];
  #pragma unroll
  for (int g = 0; g < 4; ++g) {
    int n = g * HID + wave * 16 + lane15;
    bv1[g] = bias1[n];
    bv2[g] = bias2[n];
  }

  v8f c1 = {0.f,0.f,0.f,0.f,0.f,0.f,0.f,0.f};
  v8f c2 = {0.f,0.f,0.f,0.f,0.f,0.f,0.f,0.f};

  __syncthreads();

  for (int t = 0; t < T; ++t) {
    // ---- stage x chunk (16 rows x 256 contiguous floats) into A1 as bf16 ----
    {
      int row = tid >> 5;               // 0..15
      int c8  = (tid & 31) * 8;         // 0..248
      const float* src = input + ((size_t)(r0 + row) * NP + t) * SEG + c8;
      float4 x0 = *(const float4*)src;
      float4 x1 = *(const float4*)(src + 4);
      union { __bf16 h[8]; uint4 q; } pk;
      pk.h[0] = (__bf16)x0.x; pk.h[1] = (__bf16)x0.y;
      pk.h[2] = (__bf16)x0.z; pk.h[3] = (__bf16)x0.w;
      pk.h[4] = (__bf16)x1.x; pk.h[5] = (__bf16)x1.y;
      pk.h[6] = (__bf16)x1.z; pk.h[7] = (__bf16)x1.w;
      *(uint4*)&A1[row][c8] = pk.q;
    }
    __syncthreads();

    // ---- layer 1: gates = [x|h1] W1^T + b ----
    v8f acc[4];
    gemm_gates(A1, pW1, bv1, wave, lane, acc);

    v8f h1v;
    #pragma unroll
    for (int e = 0; e < 8; ++e) {
      float iv = sigf(acc[0][e]);
      float fv = sigf(acc[1][e]);
      float gv = fast_tanh(acc[2][e]);
      float ov = sigf(acc[3][e]);
      float cn = fv * c1[e] + iv * gv;
      c1[e]  = cn;
      h1v[e] = ov * fast_tanh(cn);
    }
    __syncthreads();   // all waves finished reading A1/A2 before h1 overwrite
    {
      int u = wave * 16 + lane15;
      #pragma unroll
      for (int e = 0; e < 8; ++e) {
        int m = e + hiHalf * 8;
        __bf16 hb = (__bf16)h1v[e];
        A1[m][HID + u] = hb;   // h1 for next step's layer-1 GEMM
        A2[m][u]       = hb;   // h1 input of layer 2
      }
    }
    __syncthreads();

    // ---- layer 2: gates = [h1|h2] W2^T + b ----
    gemm_gates(A2, pW2, bv2, wave, lane, acc);

    v8f h2v;
    #pragma unroll
    for (int e = 0; e < 8; ++e) {
      float iv = sigf(acc[0][e]);
      float fv = sigf(acc[1][e]);
      float gv = fast_tanh(acc[2][e]);
      float ov = sigf(acc[3][e]);
      float cn = fv * c2[e] + iv * gv;
      c2[e]  = cn;
      h2v[e] = ov * fast_tanh(cn);
    }
    __syncthreads();
    {
      int u = wave * 16 + lane15;
      #pragma unroll
      for (int e = 0; e < 8; ++e)
        A2[e + hiHalf * 8][HID + u] = (__bf16)h2v[e];
    }
    __syncthreads();

    // ---- out[:, t] = h2 . w_lin + b_lin ----
    if (tid < 256) {
      int row = tid & 15, grp = tid >> 4;
      float s = 0.f;
      #pragma unroll
      for (int j = 0; j < 16; ++j) {
        int u = grp * 16 + j;
        s += (float)A2[row][HID + u] * wl[u];
      }
      red[grp][row] = s;
    }
    __syncthreads();
    if (tid < 16) {
      float s = blin;
      #pragma unroll
      for (int g2 = 0; g2 < 16; ++g2) s += red[g2][tid];
      out[(size_t)(r0 + tid) * T + t] = s;
    }
    __syncthreads();
  }
}

extern "C" void kernel_launch(void* const* d_in, const int* in_sizes, int n_in,
                              void* d_out, int out_size, void* d_ws, size_t ws_size,
                              hipStream_t stream)
{
  const float* input = (const float*)d_in[0];
  const float* w_ih1 = (const float*)d_in[1];
  const float* w_hh1 = (const float*)d_in[2];
  const float* b_ih1 = (const float*)d_in[3];
  const float* b_hh1 = (const float*)d_in[4];
  const float* w_ih2 = (const float*)d_in[5];
  const float* w_hh2 = (const float*)d_in[6];
  const float* b_ih2 = (const float*)d_in[7];
  const float* b_hh2 = (const float*)d_in[8];
  const float* w_lin = (const float*)d_in[9];
  const float* b_lin = (const float*)d_in[10];

  __bf16* pW1  = (__bf16*)d_ws;
  __bf16* pW2  = pW1 + PW_ELEMS;
  float*  bias1 = (float*)(pW2 + PW_ELEMS);
  float*  bias2 = bias1 + 4 * HID;

  int NP = in_sizes[0] / (BATCH * SEG);
  int T  = out_size / BATCH;

  pack_kernel<<<4096, 256, 0, stream>>>(w_ih1, w_hh1, b_ih1, b_hh1,
                                        w_ih2, w_hh2, b_ih2, b_hh2,
                                        pW1, pW2, bias1, bias2);
  lstm_main<<<NBLK, THREADS, 0, stream>>>(input, pW1, pW2, bias1, bias2,
                                          w_lin, b_lin, (float*)d_out, T, NP);
}
